// SAINTN_26147760898556
// MI455X (gfx1250) — matrix operand order repaired and checked by reading, not compiled, not run on hardware
//
#include <hip/hip_runtime.h>
#include <hip/hip_bf16.h>

typedef __attribute__((ext_vector_type(16))) __bf16 v16bf;
typedef __attribute__((ext_vector_type(8)))  __bf16 v8bf;
typedef __attribute__((ext_vector_type(8)))  float  v8f;

union BF16x16 { v16bf v; v8bf h[2]; };

// ---------------------------------------------------------------------------
// Degree / norm kernels
// ---------------------------------------------------------------------------
__global__ void k_deg_init(float* __restrict__ deg, int n) {
    int i = blockIdx.x * blockDim.x + threadIdx.x;
    if (i < n) deg[i] = 1.0f;                     // self-loop contributes 1
}

__global__ void k_deg_edges(float* __restrict__ deg, const int* __restrict__ dst,
                            long long e) {
    long long i = (long long)blockIdx.x * blockDim.x + threadIdx.x;
    if (i < e) atomicAdd(&deg[dst[i]], 1.0f);
}

__global__ void k_rsqrt_inplace(float* __restrict__ deg, int n) {
    int i = blockIdx.x * blockDim.x + threadIdx.x;
    if (i < n) deg[i] = rsqrtf(deg[i]);           // deg >= 1 always
}

// ---------------------------------------------------------------------------
// One 32-wide K step: build A fragment (ISA 16x32 bf16 layout), then 4 WMMAs
// against B fragments pulled from LDS (transposed W, col stride colStride).
// TAIL=false: Krem >= 32 guaranteed -> 4 unconditional float4 loads, no selects.
// TAIL=true : clamped float2 loads + value-only selects (runs at most once).
// ---------------------------------------------------------------------------
template <bool TAIL>
__device__ __forceinline__ void gemm_step(const float* __restrict__ Ap, int Krem,
                                          const __bf16* __restrict__ sW, int boff,
                                          int kbA, int kbB, int nlane, int colStride,
                                          v8f acc[4]) {
    v16bf a;
    if (!TAIL) {
        // Lane needs k = kbA..kbA+7 and kbA+16..kbA+23 : two 16B-aligned 32B chunks.
        float4 q0 = *reinterpret_cast<const float4*>(Ap + kbA);
        float4 q1 = *reinterpret_cast<const float4*>(Ap + kbA + 4);
        float4 q2 = *reinterpret_cast<const float4*>(Ap + kbA + 16);
        float4 q3 = *reinterpret_cast<const float4*>(Ap + kbA + 20);
        a[0]  = (__bf16)q0.x; a[1]  = (__bf16)q0.y; a[2]  = (__bf16)q0.z; a[3]  = (__bf16)q0.w;
        a[4]  = (__bf16)q1.x; a[5]  = (__bf16)q1.y; a[6]  = (__bf16)q1.z; a[7]  = (__bf16)q1.w;
        a[8]  = (__bf16)q2.x; a[9]  = (__bf16)q2.y; a[10] = (__bf16)q2.z; a[11] = (__bf16)q2.w;
        a[12] = (__bf16)q3.x; a[13] = (__bf16)q3.y; a[14] = (__bf16)q3.z; a[15] = (__bf16)q3.w;
    } else {
#pragma unroll
        for (int i = 0; i < 8; ++i) {
            int koff = (i < 4) ? (2 * i) : (2 * i + 8);
            int kk   = kbA + koff;                         // even, in [0,31]
            int kk_c = (kk + 2 <= Krem) ? kk : (Krem - 2); // clamp to valid pair
            float2 f = *reinterpret_cast<const float2*>(Ap + kk_c);
            float f0 = (kk     < Krem) ? f.x : 0.0f;
            float f1 = (kk + 1 < Krem) ? f.y : 0.0f;
            a[2 * i]     = (__bf16)f0;
            a[2 * i + 1] = (__bf16)f1;
        }
    }
#pragma unroll
    for (int ct = 0; ct < 4; ++ct) {
        int off = (ct * 16 + nlane) * colStride + boff + kbB;
        BF16x16 bu;
        bu.h[0] = *reinterpret_cast<const v8bf*>(&sW[off]);
        bu.h[1] = *reinterpret_cast<const v8bf*>(&sW[off + 16]);
        acc[ct] = __builtin_amdgcn_wmma_f32_16x16x32_bf16(
            false, a, false, bu.v, (short)0, acc[ct], false, false);
    }
}

// ---------------------------------------------------------------------------
// Generic WMMA GEMM: out[N,64] = concat(A0[N,K0], A1[N,K1]) @ W[K0+K1,64] (+bias)
// Block = 128 threads (4 waves). Each wave -> 16 rows x 64 cols.
// Requirements (met by all call sites): K0,K1 even; K0 % 32 == 0 when K1 > 0;
// row strides K0,K1 give 16B-aligned rows (K0*4 % 16 == 0).
// ---------------------------------------------------------------------------
__global__ void k_gemm_wmma_bf16(const float* __restrict__ A0,
                                 const float* __restrict__ A1,
                                 int K0, int K1, int Kpad,
                                 const float* __restrict__ W,
                                 const float* __restrict__ bias,
                                 float* __restrict__ out, int Nrows) {
    extern __shared__ v8bf smem_v8[];
    __bf16* sW = reinterpret_cast<__bf16*>(smem_v8);

    const int Kreal     = K0 + K1;
    const int colStride = Kpad + 8;               // keeps 16B align, spreads LDS banks
    const int tid  = threadIdx.x;
    const int lane = tid & 31;
    const int wave = tid >> 5;

    // Stage W^T (bf16) into LDS: sW[col*colStride + k] = W[k*64 + col].
    {
        const int col = tid & 63;
        const int kst = tid >> 6;                 // 0 or 1
        __bf16* dstp = sW + col * colStride;
        for (int k = kst; k < Kpad; k += 2) {
            float v = (k < Kreal) ? W[(size_t)k * 64 + col] : 0.0f;
            dstp[k] = (__bf16)v;
        }
    }
    __syncthreads();

    const int m0     = blockIdx.x * 64 + wave * 16;
    const int half   = lane >> 4;                 // 0 or 1
    const int nlane  = lane & 15;
    const int arow   = m0 + nlane;                // A-fragment row for this lane
    const int arow_c = (arow < Nrows) ? arow : (Nrows - 1);   // clamp: always valid
    const int kbA    = half * 8;                  // A K-base per ISA layout
    const int kbB    = half * 8;                  // B K-base per ISA layout

    v8f zero = {0.f, 0.f, 0.f, 0.f, 0.f, 0.f, 0.f, 0.f};
    v8f acc[4];
#pragma unroll
    for (int ct = 0; ct < 4; ++ct) acc[ct] = zero;

    // ---- A0 region: full steps, then optional tail ----
    {
        const float* base0 = A0 + (size_t)arow_c * K0;
        const int full0 = K0 & ~31;
        for (int ks = 0; ks < full0; ks += 32)
            gemm_step<false>(base0 + ks, 32, sW, ks, kbA, kbB, nlane, colStride, acc);
        if (K0 > full0)
            gemm_step<true>(base0 + full0, K0 - full0, sW, full0, kbA, kbB, nlane, colStride, acc);
    }
    // ---- A1 region (concat case; K0 % 32 == 0 guaranteed here) ----
    if (K1 > 0) {
        const float* base1 = A1 + (size_t)arow_c * K1;
        const int b1off = (K0 + 31) & ~31;
        const int full1 = K1 & ~31;
        for (int ks = 0; ks < full1; ks += 32)
            gemm_step<false>(base1 + ks, 32, sW, b1off + ks, kbA, kbB, nlane, colStride, acc);
        if (K1 > full1)
            gemm_step<true>(base1 + full1, K1 - full1, sW, b1off + full1, kbA, kbB, nlane, colStride, acc);
    }

    // Epilogue: D layout -> lane holds (M = r + 8*half, N = nlane) in VGPR r
#pragma unroll
    for (int ct = 0; ct < 4; ++ct) {
        int col = ct * 16 + nlane;
        float bv = bias ? bias[col] : 0.0f;
#pragma unroll
        for (int r = 0; r < 8; ++r) {
            int row = m0 + r + 8 * half;
            if (row < Nrows) out[(size_t)row * 64 + col] = acc[ct][r] + bv;
        }
    }
}

// ---------------------------------------------------------------------------
// Message passing: pre = dinv^2 * t (self-loop), then scatter-add edges,
// then bias + ReLU.
// ---------------------------------------------------------------------------
__global__ void k_self_init(float* __restrict__ pre, const float* __restrict__ t,
                            const float* __restrict__ dinv, int n) {
    long long idx = (long long)blockIdx.x * blockDim.x + threadIdx.x;
    long long tot = (long long)n * 64;
    if (idx < tot) {
        int i = (int)(idx >> 6);
        float w = dinv[i] * dinv[i];
        pre[idx] = w * t[idx];
    }
}

__global__ void k_scatter_edges(float* __restrict__ pre, const float* __restrict__ t,
                                const float* __restrict__ dinv,
                                const int* __restrict__ src, const int* __restrict__ dst,
                                long long e) {
    long long idx = (long long)blockIdx.x * blockDim.x + threadIdx.x;
    long long tot = e << 6;
    if (idx < tot) {
        long long ed = idx >> 6;
        int f = (int)(idx & 63);
        int s = src[ed], d = dst[ed];
        float w = dinv[s] * dinv[d];
        atomicAdd(&pre[(size_t)d * 64 + f], w * t[(size_t)s * 64 + f]);
    }
}

__global__ void k_bias_relu(float* __restrict__ x, const float* __restrict__ b, int n) {
    long long idx = (long long)blockIdx.x * blockDim.x + threadIdx.x;
    long long tot = (long long)n * 64;
    if (idx < tot) {
        int f = (int)(idx & 63);
        x[idx] = fmaxf(x[idx] + b[f], 0.0f);
    }
}

// ---------------------------------------------------------------------------
// log_softmax over 64 classes: one wave32 per row, 2 elements per lane.
// ---------------------------------------------------------------------------
__global__ void k_log_softmax64(float* __restrict__ out, int n) {
    int gtid = blockIdx.x * blockDim.x + threadIdx.x;
    int row  = gtid >> 5;
    int lane = threadIdx.x & 31;
    if (row >= n) return;
    size_t base = (size_t)row * 64;
    float v0 = out[base + lane];
    float v1 = out[base + lane + 32];
    float m = fmaxf(v0, v1);
#pragma unroll
    for (int off = 16; off; off >>= 1) m = fmaxf(m, __shfl_xor(m, off, 32));
    float s = expf(v0 - m) + expf(v1 - m);
#pragma unroll
    for (int off = 16; off; off >>= 1) s += __shfl_xor(s, off, 32);
    float l = logf(s);
    out[base + lane]      = v0 - m - l;
    out[base + lane + 32] = v1 - m - l;
}

// ---------------------------------------------------------------------------
// Driver
// ---------------------------------------------------------------------------
extern "C" void kernel_launch(void* const* d_in, const int* in_sizes, int n_in,
                              void* d_out, int out_size, void* d_ws, size_t ws_size,
                              hipStream_t stream) {
    const float* x    = (const float*)d_in[0];
    const int*   eidx = (const int*)  d_in[1];
    const float* W1   = (const float*)d_in[2];
    const float* b1   = (const float*)d_in[3];
    const float* W2   = (const float*)d_in[4];
    const float* b2   = (const float*)d_in[5];
    const float* Wl   = (const float*)d_in[6];
    const float* bl   = (const float*)d_in[7];

    const int HID = in_sizes[3];                  // 64
    const int FIN = in_sizes[2] / HID;            // 500
    const int N   = in_sizes[0] / FIN;            // 100000
    const long long E = in_sizes[1] / 2;          // 1600000
    const int* srcI = eidx;
    const int* dstI = eidx + E;

    // Workspace layout
    auto align256 = [](size_t v) { return (v + 255) & ~(size_t)255; };
    char* ws = (char*)d_ws;
    size_t o = 0;
    float* dinv = (float*)(ws + o); o += align256((size_t)N * sizeof(float));
    float* t    = (float*)(ws + o); o += align256((size_t)N * 64 * sizeof(float));
    float* p1   = (float*)(ws + o); o += align256((size_t)N * 64 * sizeof(float));
    float* p2   = (float*)(ws + o); o += align256((size_t)N * 64 * sizeof(float));
    (void)ws_size; (void)n_in; (void)out_size;

    const int T = 256;
    long long nf  = (long long)N * 64;
    long long ef  = E * 64;
    unsigned bN   = (unsigned)((N + T - 1) / T);
    unsigned bE   = (unsigned)((E + T - 1) / T);
    unsigned bNF  = (unsigned)((nf + T - 1) / T);
    unsigned bEF  = (unsigned)((ef + T - 1) / T);
    unsigned bG   = (unsigned)((N + 63) / 64);          // 64 rows per GEMM block
    unsigned bSM  = (unsigned)(((long long)N * 32 + T - 1) / T);

    // Normalization: deg (with self loop) -> dinv = rsqrt(deg)
    k_deg_init     <<<bN, T, 0, stream>>>(dinv, N);
    k_deg_edges    <<<bE, T, 0, stream>>>(dinv, dstI, E);
    k_rsqrt_inplace<<<bN, T, 0, stream>>>(dinv, N);

    // Layer 1: t = x @ W1 ; p1 = scatter(norm * t) ; x1 = relu(p1 + b1)
    int Kpad1 = ((FIN + 31) / 32) * 32;                 // 512
    size_t sm1 = (size_t)64 * (Kpad1 + 8) * sizeof(__bf16);
    k_gemm_wmma_bf16<<<bG, 128, sm1, stream>>>(x, x, FIN, 0, Kpad1, W1, nullptr, t, N);
    k_self_init    <<<bNF, T, 0, stream>>>(p1, t, dinv, N);
    k_scatter_edges<<<bEF, T, 0, stream>>>(p1, t, dinv, srcI, dstI, E);
    k_bias_relu    <<<bNF, T, 0, stream>>>(p1, b1, N);

    // Layer 2: t = x1 @ W2 ; p2 = scatter(norm * t) ; x2 = relu(p2 + b2)
    size_t sm2 = (size_t)64 * (64 + 8) * sizeof(__bf16);
    k_gemm_wmma_bf16<<<bG, 128, sm2, stream>>>(p1, p1, 64, 0, 64, W2, nullptr, t, N);
    k_self_init    <<<bNF, T, 0, stream>>>(p2, t, dinv, N);
    k_scatter_edges<<<bEF, T, 0, stream>>>(p2, t, dinv, srcI, dstI, E);
    k_bias_relu    <<<bNF, T, 0, stream>>>(p2, b2, N);

    // Final linear on concat([x1,x2]) + log_softmax (in place in d_out)
    float* logits = (float*)d_out;
    size_t sm3 = (size_t)64 * (128 + 8) * sizeof(__bf16);
    k_gemm_wmma_bf16<<<bG, 128, sm3, stream>>>(p1, p2, 64, 64, 128, Wl, bl, logits, N);
    k_log_softmax64<<<bSM, T, 0, stream>>>(logits, N);
}
